// GaussianVideo3D2D_49890340110844
// MI455X (gfx1250) — compile-verified, matrix operand order checked by reading, not run on hardware
//
#include <hip/hip_runtime.h>

typedef _Float16 v16h __attribute__((ext_vector_type(16)));
typedef float    v8f  __attribute__((ext_vector_type(8)));

#define NG   1024
#define W_   128
#define H_   128
#define T_   4
#define NFROW 5  // feature rows in LDS: 0..2 = channels, 3..4 = zeros

// s = -0.5 / ln(2): folds exp(-0.5*q) == exp2(s*q) into the Sinv coefficients.
#define EXP2_SCALE (-0.72134752044448170368f)

// ---------------------------------------------------------------------------
// Kernel 1: per-gaussian prep.
//   means = tanh(xyz); alpha = sigmoid(opac)
//   L = SCALE * tril(chol + bound);  Sinv = L^-T L^-1 (analytic)
// Stores the six quadratic-form coefficients PRE-SCALED by s = -0.5/ln2 and
// la = log2(alpha). Features are written channel-major (transposed) as f16:
// f16t[c*NG + g], rows 3..4 zeroed (zero-columns for WMMA B lanes n >= 3).
// ---------------------------------------------------------------------------
__global__ __launch_bounds__(256) void prep_kernel(
    const float* __restrict__ xyz, const float* __restrict__ chol,
    const float* __restrict__ feats, const float* __restrict__ opac,
    float* __restrict__ params, _Float16* __restrict__ f16t) {
  int g = blockIdx.x * blockDim.x + threadIdx.x;
  if (g >= NG) return;
  const float SCALE = 2.0f * 16.0f / 128.0f;  // 0.25

  float mx = tanhf(xyz[g * 3 + 0]);
  float my = tanhf(xyz[g * 3 + 1]);
  float mt = tanhf(xyz[g * 3 + 2]);

  float l00 = (chol[g * 6 + 0] + 0.5f) * SCALE;
  float l10 = (chol[g * 6 + 1]) * SCALE;
  float l20 = (chol[g * 6 + 2]) * SCALE;
  float l11 = (chol[g * 6 + 3] + 0.5f) * SCALE;
  float l21 = (chol[g * 6 + 4]) * SCALE;
  float l22 = (chol[g * 6 + 5] + 0.5f) * SCALE;

  float i00 = 1.0f / l00, i11 = 1.0f / l11, i22 = 1.0f / l22;
  float i10 = -l10 * i00 * i11;
  float i21 = -l21 * i11 * i22;
  float i20 = (l10 * l21 - l11 * l20) * (i00 * i11 * i22);

  const float s = EXP2_SCALE;
  float a_s  = s * (i00 * i00 + i10 * i10 + i20 * i20);   // s*Sinv00
  float ab_s = s * 2.0f * (i10 * i11 + i20 * i21);        // s*2*Sinv01
  float at_s = s * 2.0f * (i20 * i22);                    // s*2*Sinv02
  float b_s  = s * (i11 * i11 + i21 * i21);               // s*Sinv11
  float bt_s = s * 2.0f * (i21 * i22);                    // s*2*Sinv12
  float c_s  = s * (i22 * i22);                           // s*Sinv22

  float alpha = 1.0f / (1.0f + __expf(-opac[g]));
  float la = log2f(alpha);

  float* p = params + g * 12;
  p[0] = a_s;  p[1] = b_s;  p[2] = c_s;
  p[3] = ab_s; p[4] = at_s; p[5] = bt_s;
  p[6] = mx;   p[7] = my;   p[8] = mt;
  p[9] = la;   p[10] = 0.0f; p[11] = 0.0f;

  f16t[0 * NG + g] = (_Float16)feats[g * 3 + 0];
  f16t[1 * NG + g] = (_Float16)feats[g * 3 + 1];
  f16t[2 * NG + g] = (_Float16)feats[g * 3 + 2];
  f16t[3 * NG + g] = (_Float16)0.0f;
  f16t[4 * NG + g] = (_Float16)0.0f;
}

// ---------------------------------------------------------------------------
// Kernel 2: one block per (t, h) row (512 blocks); all 8 waves share (y, tc),
// so the exponent is a per-gaussian quadratic in x:
//   expo(x) = ca*x^2 + cb*x + cc   (alpha and -0.5/ln2 pre-folded)
// Block prologue computes (ca, cb, cc) for all 1024 gaussians into LDS.
// Hot loop per wave: K=1024 in 32-gaussian WMMA steps; per density just
// 2 FMA + 1 v_exp_f32; B operand fetched as 2x ds_load_b128 from the
// channel-major f16 feature rows.
//
// 16-bit A-matrix per-lane K layout (ISA 7.12.2):
//   lanes 0-15  (M = lane):    elems 0..7 -> K 0..7,  elems 8..15 -> K 16..23
//   lanes 16-31 (M = lane-16): elems 0..7 -> K 8..15, elems 8..15 -> K 24..31
// B mirrored (lane -> N column, same per-lane K indexing).
// ---------------------------------------------------------------------------
__global__ __launch_bounds__(256) void splat_kernel(
    const float* __restrict__ params, const _Float16* __restrict__ f16t,
    float* __restrict__ out) {
  __shared__ __align__(16) float4   sC[NG];           // 16 KB: ca, cb, cc, pad
  __shared__ __align__(16) _Float16 sFt[NFROW * NG];  // 10 KB channel-major

  const int tid = threadIdx.x;
  const int row = blockIdx.x;          // t*H + h, 0..511
  const int h = row & (H_ - 1);
  const int t = row >> 7;
  const float y  = ((float)h + 0.5f) * (2.0f / (float)H_) - 1.0f;
  const float tc = ((float)t + 0.5f) * (2.0f / (float)T_) - 1.0f;

  {  // stage transposed features (incl. zero rows): flat 10 KB copy
    uint4* d1 = (uint4*)sFt;
    const uint4* s1 = (const uint4*)f16t;
    for (int i = tid; i < NFROW * NG * 2 / 16; i += 256) d1[i] = s1[i];
  }

  // per-(gaussian,row) quadratic coefficients
  for (int g = tid; g < NG; g += 256) {
    const float4* gp = (const float4*)(params + g * 12);
    const float4 p0 = gp[0];  // a_s, b_s, c_s, ab_s
    const float4 p1 = gp[1];  // at_s, bt_s, mx, my
    const float4 p2 = gp[2];  // mt, la, 0, 0
    const float mx = p1.z;
    const float dy = y - p1.w;
    const float dt = tc - p2.x;
    const float txx = p0.x * mx;                  // a_s*mx
    const float u   = fmaf(p1.x, dt, p0.w * dy);  // at_s*dt + ab_s*dy
    const float cb  = fmaf(-2.0f, txx, u);        // x-linear coefficient
    const float v   = fmaf(p1.y, dt, p0.y * dy);  // b_s*dy + bt_s*dt
    const float w2  = p0.z * dt;                  // c_s*dt
    float cc = fmaf(mx, txx - u, p2.y);           // la + mx*(a_s*mx - u)
    cc = fmaf(dy, v, cc);
    cc = fmaf(dt, w2, cc);
    sC[g] = make_float4(p0.x, cb, cc, 0.0f);
  }
  __syncthreads();

  const int lane = tid & 31;
  const int wave = tid >> 5;
  const int m = lane & 15;                  // A-matrix row
  const int wpix = wave * 16 + m;           // w coordinate of this lane
  const float x = ((float)wpix + 0.5f) * (2.0f / (float)W_) - 1.0f;

  const int koff = (lane & 16) ? 8 : 0;     // per-half-wave K base
  const int n = lane & 15;                  // B/D column (channel)
  const int fb = ((n < 4) ? n : 4) * NG;    // feature row base (zero rows 3/4)

  v8f acc = {};

#pragma unroll 2
  for (int kb = 0; kb < NG; kb += 32) {
    const int g0 = kb + koff;

    // B: two contiguous 16B runs of f16 feats -> 2x ds_load_b128
    union { v16h h; uint4 q[2]; } B;
    B.q[0] = *(const uint4*)&sFt[fb + g0];
    B.q[1] = *(const uint4*)&sFt[fb + g0 + 16];

    v16h A;
#pragma unroll
    for (int j = 0; j < 16; ++j) {
      const int g = g0 + (j < 8 ? j : j + 8);
      const float4 c = sC[g];
      const float expo = fmaf(x, fmaf(c.x, x, c.y), c.z);
      A[j] = (_Float16)__builtin_amdgcn_exp2f(expo);
    }
    acc = __builtin_amdgcn_wmma_f32_16x16x32_f16(
        false, A, false, B.h, (short)0, acc, false, false);
  }

  // D layout: VGPR i -> (M=i, N=lane) lanes 0-15, (M=i+8, N=lane-16) 16-31.
  if (n < 3) {
    const int mb = (lane & 16) ? 8 : 0;
#pragma unroll
    for (int i = 0; i < 8; ++i) {
      const int pix = row * W_ + wave * 16 + mb + i;  // flat (t*H+h)*W + w
      float vv = acc[i];
      vv = fminf(fmaxf(vv, 0.0f), 1.0f);
      out[pix * 3 + n] = vv;
    }
  }
}

extern "C" void kernel_launch(void* const* d_in, const int* in_sizes, int n_in,
                              void* d_out, int out_size, void* d_ws, size_t ws_size,
                              hipStream_t stream) {
  const float* xyz   = (const float*)d_in[0];  // (1024, 3)
  const float* chol  = (const float*)d_in[1];  // (1024, 6)
  const float* feats = (const float*)d_in[2];  // (1024, 3)
  const float* opac  = (const float*)d_in[3];  // (1024, 1)

  float*    params = (float*)d_ws;                                     // 48 KB
  _Float16* f16t = (_Float16*)((char*)d_ws + NG * 12 * sizeof(float)); // 10 KB

  prep_kernel<<<(NG + 255) / 256, 256, 0, stream>>>(xyz, chol, feats, opac,
                                                    params, f16t);
  splat_kernel<<<T_ * H_, 256, 0, stream>>>(params, f16t, (float*)d_out);
}
